// MultiHeadAttention_15564961480985
// MI455X (gfx1250) — compile-verified, matrix operand order checked by reading, not run on hardware
//
#include <hip/hip_runtime.h>
#include <hip/hip_bf16.h>

// ---------------------------------------------------------------------------
// MI455X (gfx1250) fused multi-head attention, bf16 WMMA + flash softmax.
//   prep:   X(q/k/v) f32->bf16; W(q/k/v/o) f32->bf16 transposed [n][k]
//   pass:   q/k/v = X @ W + b      (bf16 WMMA GEMM, async-LDS staging)
//   pass:   flash attention        (QK^T via TDM-staged K, online softmax, PV)
//   pass:   out = attn @ Wo + bo   (bf16 WMMA GEMM, f32 out)
// Logits never touch HBM (would add >1 GB of traffic at S=2048).
// ---------------------------------------------------------------------------

typedef __attribute__((ext_vector_type(16))) __bf16 v16bf;
typedef __attribute__((ext_vector_type(8)))  float  v8f;
typedef unsigned int v4u __attribute__((ext_vector_type(4)));
typedef int          v8i __attribute__((ext_vector_type(8)));
typedef int          v4i __attribute__((ext_vector_type(4)));

#ifndef __has_builtin
#define __has_builtin(x) 0
#endif
#if __has_builtin(__builtin_amdgcn_tensor_load_to_lds) && __has_builtin(__builtin_amdgcn_s_wait_tensorcnt)
#define HAVE_TDM 1
#else
#define HAVE_TDM 0
#endif
#if __has_builtin(__builtin_amdgcn_global_load_async_to_lds_b128) && __has_builtin(__builtin_amdgcn_s_wait_asynccnt)
#define HAVE_ASYNC 1
#else
#define HAVE_ASYNC 0
#endif

// generic -> explicit address-space typed pointers for the async builtin:
// signature is (v4i AS1* src, v4i AS3* dst, imm offset, imm cpol).
#define AS1P(p) ((__attribute__((address_space(1))) v4i*)(__UINTPTR_TYPE__)(p))
#define AS3P(p) ((__attribute__((address_space(3))) v4i*)(unsigned int)(__UINTPTR_TYPE__)(p))

union FragU {
  v16bf v;
  uint4 u[2];
};

__device__ __forceinline__ unsigned short f32_to_bf16(float f) {
  union { float f; unsigned int u; } c; c.f = f;
  unsigned int u = c.u + 0x7FFFu + ((c.u >> 16) & 1u);  // round-nearest-even
  return (unsigned short)(u >> 16);
}

#define WMMA_BF16(A, Bm, C) \
  __builtin_amdgcn_wmma_f32_16x16x32_bf16(false, (A), false, (Bm), (short)0, (C), false, false)

// ---------------------------------------------------------------------------
// Prep kernels: one-time f32->bf16 conversion so GEMM hot loops are pure copies.
// ---------------------------------------------------------------------------
__global__ __launch_bounds__(256) void cvt_bf16_kernel(
    const float* __restrict__ in, unsigned short* __restrict__ out, int n4) {
  int i = blockIdx.x * 256 + threadIdx.x;
  if (i < n4) {
    float4 d = ((const float4*)in)[i];
    ushort4 h;
    h.x = f32_to_bf16(d.x); h.y = f32_to_bf16(d.y);
    h.z = f32_to_bf16(d.z); h.w = f32_to_bf16(d.w);
    ((ushort4*)out)[i] = h;
  }
}

// Wt[n][k] = bf16(W[k][n]) ; 32x32 LDS tile, coalesced both sides.
__global__ __launch_bounds__(256) void wt_kernel(
    const float* __restrict__ W, unsigned short* __restrict__ Wt) {
  __shared__ unsigned short t[32][33];
  const int bn = blockIdx.x * 32;   // n tile
  const int bk = blockIdx.y * 32;   // k tile
  const int x = threadIdx.x;        // 0..31
  #pragma unroll
  for (int i = 0; i < 4; ++i) {
    int y = threadIdx.y * 4 + i;
    t[y][x] = f32_to_bf16(W[(size_t)(bk + y) * 1024 + bn + x]);
  }
  __syncthreads();
  #pragma unroll
  for (int i = 0; i < 4; ++i) {
    int y = threadIdx.y * 4 + i;
    Wt[(size_t)(bn + y) * 1024 + bk + x] = t[x][y];
  }
}

// ---------------------------------------------------------------------------
// GEMM: C[M=4096, N=1024] = A_bf16[M,1024] * Wt_bf16[N,1024]^T + bias
// Block tile 128x128, BK=32, 256 threads (8 waves, 4x2 grid, 32x64/wave).
// A and B tiles staged with async global->LDS b128 copies when available.
// ---------------------------------------------------------------------------
template<bool OUT_IS_F32>
__global__ __launch_bounds__(256) void gemm_bias_kernel(
    const unsigned short* __restrict__ A, const unsigned short* __restrict__ Wt,
    const float* __restrict__ bias, void* __restrict__ Cptr)
{
  constexpr int Kdim = 1024, Ndim = 1024;
  constexpr int BK = 32, LDT = 40;  // +8 bf16 pad -> conflict-free rows
  __shared__ unsigned short As[2][128 * LDT];
  __shared__ unsigned short Bs[2][128 * LDT];   // rows of Wt: [n][k]

  const int tid  = threadIdx.x;
  const int lane = tid & 31;
  const int wid  = tid >> 5;
  const int half = lane >> 4;
  const int lrow = lane & 15;
  const int wm   = wid & 3;
  const int wn   = wid >> 2;
  const int bm   = blockIdx.y * 128;
  const int bn   = blockIdx.x * 128;

  v8f acc[2][4];
  #pragma unroll
  for (int mt = 0; mt < 2; ++mt)
    #pragma unroll
    for (int nt = 0; nt < 4; ++nt)
      #pragma unroll
      for (int e = 0; e < 8; ++e) acc[mt][nt][e] = 0.0f;

  auto stage = [&](int buf, int kt) {
    const int k0 = kt * BK;
    #pragma unroll
    for (int i = 0; i < 2; ++i) {
      int idx = tid + i * 256;
      int r = idx >> 2;            // 4 x 16B chunks per 32-elem row
      int c = (idx & 3) * 8;
      const unsigned short* ga = A  + (size_t)(bm + r) * Kdim + k0 + c;
      const unsigned short* gb = Wt + (size_t)(bn + r) * Kdim + k0 + c;
#if HAVE_ASYNC
      __builtin_amdgcn_global_load_async_to_lds_b128(AS1P(ga), AS3P(&As[buf][r * LDT + c]), 0, 0);
      __builtin_amdgcn_global_load_async_to_lds_b128(AS1P(gb), AS3P(&Bs[buf][r * LDT + c]), 0, 0);
#else
      *(uint4*)(&As[buf][r * LDT + c]) = *(const uint4*)ga;
      *(uint4*)(&Bs[buf][r * LDT + c]) = *(const uint4*)gb;
#endif
      if (k0 + BK < Kdim) {        // prefetch two k-steps ahead of compute
        __builtin_prefetch(ga + BK, 0, 0);
        __builtin_prefetch(gb + BK, 0, 0);
      }
    }
  };

  stage(0, 0);
#if HAVE_ASYNC
  __builtin_amdgcn_s_wait_asynccnt(0);
#endif
  __syncthreads();
  constexpr int KT = Kdim / BK;
  for (int kt = 0; kt < KT; ++kt) {
    const int buf = kt & 1;
    if (kt + 1 < KT) stage(buf ^ 1, kt + 1);

    FragU a[2], bfr[4];
    #pragma unroll
    for (int mt = 0; mt < 2; ++mt) {
      const unsigned short* p = &As[buf][(wm * 32 + mt * 16 + lrow) * LDT + 8 * half];
      a[mt].u[0] = *(const uint4*)p;
      a[mt].u[1] = *(const uint4*)(p + 16);
    }
    #pragma unroll
    for (int nt = 0; nt < 4; ++nt) {
      const unsigned short* p = &Bs[buf][(wn * 64 + nt * 16 + lrow) * LDT + 8 * half];
      bfr[nt].u[0] = *(const uint4*)p;
      bfr[nt].u[1] = *(const uint4*)(p + 16);
    }
    #pragma unroll
    for (int mt = 0; mt < 2; ++mt)
      #pragma unroll
      for (int nt = 0; nt < 4; ++nt)
        acc[mt][nt] = WMMA_BF16(a[mt].v, bfr[nt].v, acc[mt][nt]);
#if HAVE_ASYNC
    __builtin_amdgcn_s_wait_asynccnt(0);
#endif
    __syncthreads();
  }

  #pragma unroll
  for (int mt = 0; mt < 2; ++mt)
    #pragma unroll
    for (int nt = 0; nt < 4; ++nt) {
      const int col = bn + wn * 64 + nt * 16 + lrow;
      const float bv = bias[col];
      #pragma unroll
      for (int v = 0; v < 8; ++v) {
        const int row = bm + wm * 32 + mt * 16 + v + 8 * half;
        const float val = acc[mt][nt][v] + bv;
        if (OUT_IS_F32) ((float*)Cptr)[(size_t)row * Ndim + col] = val;
        else ((unsigned short*)Cptr)[(size_t)row * Ndim + col] = f32_to_bf16(val);
      }
    }
}

// ---------------------------------------------------------------------------
// Flash attention: grid (B*H, S/256), 256 threads = 8 waves, 32 q-rows/wave.
// K tile staged by the Tensor Data Mover (one DMA per 64x64 bf16 tile, with
// D# padding replicating the 72-ushort LDS row stride); V staged transposed.
// ---------------------------------------------------------------------------
__global__ __launch_bounds__(256) void flash_attn_kernel(
    const unsigned short* __restrict__ Qb,
    const unsigned short* __restrict__ Kb,
    const unsigned short* __restrict__ Vb,
    const float* __restrict__ mask,
    unsigned short* __restrict__ Ob)
{
  constexpr int S = 2048, D = 1024, DH = 64;
  constexpr int NK = 64;
  constexpr int LK = 72, LV = 72, LP = 72;   // padded row strides (ushorts)
  __shared__ unsigned short Ks[NK * LK];     // [key][d]
  __shared__ unsigned short Vs[DH * LV];     // [d][key]  (transposed)
  __shared__ unsigned short Ps[8][32 * LP];  // per-wave P tile [qrow][key]

  const int tid  = threadIdx.x;
  const int lane = tid & 31;
  const int wid  = tid >> 5;
  const int half = lane >> 4;
  const int lrow = lane & 15;

  const int bh = blockIdx.x;
  const int b  = bh >> 4;
  const int h  = bh & 15;
  const int q0 = blockIdx.y * 256 + wid * 32;

  const unsigned short* qp = Qb + (size_t)b * S * D + h * DH;
  const unsigned short* kp = Kb + (size_t)b * S * D + h * DH;
  const unsigned short* vp = Vb + (size_t)b * S * D + h * DH;
  const float* mp = mask + (size_t)b * S;
  unsigned short* Pw = Ps[wid];

  FragU aq[2][2];
  #pragma unroll
  for (int mt = 0; mt < 2; ++mt)
    #pragma unroll
    for (int kk = 0; kk < 2; ++kk) {
      const unsigned short* p =
          qp + (size_t)(q0 + mt * 16 + lrow) * D + kk * 32 + 8 * half;
      aq[mt][kk].u[0] = *(const uint4*)p;
      aq[mt][kk].u[1] = *(const uint4*)(p + 16);
    }

  v8f acc_o[2][4];
  float rmax[2][8], rsum[2][8];
  #pragma unroll
  for (int mt = 0; mt < 2; ++mt) {
    #pragma unroll
    for (int dn = 0; dn < 4; ++dn)
      #pragma unroll
      for (int e = 0; e < 8; ++e) acc_o[mt][dn][e] = 0.0f;
    #pragma unroll
    for (int v = 0; v < 8; ++v) { rmax[mt][v] = -1e30f; rsum[mt][v] = 0.0f; }
  }

  for (int j = 0; j < S; j += NK) {
    // --- stage K tile [key][d] ---
#if HAVE_TDM
    if (wid == 0) {
      unsigned long long ga = (unsigned long long)(__UINTPTR_TYPE__)(kp + (size_t)j * D);
      v4u g0;
      g0[0] = 1u;                                   // count=1, user descriptor
      g0[1] = (unsigned int)(__UINTPTR_TYPE__)&Ks[0];  // lds_addr (bytes)
      g0[2] = (unsigned int)(ga & 0xffffffffu);     // global_addr[31:0]
      g0[3] = (unsigned int)((ga >> 32) & 0x1ffffffu) | (2u << 30);  // addr[56:32], type=2
      v8i g1;
      g1[0] = (int)((1u << 16)      // data_size = 2 bytes
                  | (1u << 20)      // pad_enable
                  | (4u << 22)      // pad_interval = 32 DWORDs (one 128B row)
                  | (3u << 25));    // pad_amount   = 4 DWORDs  (16B -> 72-ushort stride)
      g1[1] = (int)(64u << 16);     // tensor_dim0 = 64 (bits 79:48, low half)
      g1[2] = (int)(64u << 16);     // tensor_dim1 = 64 (bits 111:80, low half)
      g1[3] = (int)(64u << 16);     // tile_dim0 = 64 (bits 127:112)
      g1[4] = 64;                   // tile_dim1 = 64 (bits 143:128); tile_dim2 = 0
      g1[5] = 1024;                 // tensor_dim0_stride = D (bits 207:160, low 32)
      g1[6] = 0;
      g1[7] = 0;
      v4i gz = {0, 0, 0, 0};
#if __clang_major__ >= 23
      v8i gz8 = {0, 0, 0, 0, 0, 0, 0, 0};
      __builtin_amdgcn_tensor_load_to_lds(g0, g1, gz, gz, gz8, 0);
#else
      __builtin_amdgcn_tensor_load_to_lds(g0, g1, gz, gz, 0);
#endif
      __builtin_amdgcn_s_wait_tensorcnt(0);
    }
#elif HAVE_ASYNC
    #pragma unroll
    for (int i = 0; i < 2; ++i) {
      int idx = tid + i * 256;
      int r = idx >> 3;
      int c = (idx & 7) * 8;
      __builtin_amdgcn_global_load_async_to_lds_b128(
          AS1P(kp + (size_t)(j + r) * D + c), AS3P(&Ks[r * LK + c]), 0, 0);
    }
#else
    #pragma unroll
    for (int i = 0; i < 2; ++i) {
      int idx = tid + i * 256;
      int r = idx >> 3;
      int c = (idx & 7) * 8;
      *(uint4*)(&Ks[r * LK + c]) = *(const uint4*)(kp + (size_t)(j + r) * D + c);
    }
#endif
    // --- stage V tile transposed [d][key] ---
    #pragma unroll
    for (int i = 0; i < 2; ++i) {
      int idx = tid + i * 256;
      int r = idx >> 3;
      int c = (idx & 7) * 8;
      uint4 d = *(const uint4*)(vp + (size_t)(j + r) * D + c);
      const unsigned short* e = (const unsigned short*)&d;
      #pragma unroll
      for (int t = 0; t < 8; ++t) Vs[(c + t) * LV + r] = e[t];
    }
#if !HAVE_TDM && HAVE_ASYNC
    __builtin_amdgcn_s_wait_asynccnt(0);
#endif
    __syncthreads();

    // --- S = Q K^T (32 x 64 logits per wave) ---
    v8f sacc[2][4];
    #pragma unroll
    for (int mt = 0; mt < 2; ++mt)
      #pragma unroll
      for (int nt = 0; nt < 4; ++nt)
        #pragma unroll
        for (int e = 0; e < 8; ++e) sacc[mt][nt][e] = 0.0f;

    #pragma unroll
    for (int kk = 0; kk < 2; ++kk) {
      FragU bk[4];
      #pragma unroll
      for (int nt = 0; nt < 4; ++nt) {
        const unsigned short* p = &Ks[(nt * 16 + lrow) * LK + kk * 32 + 8 * half];
        bk[nt].u[0] = *(const uint4*)p;
        bk[nt].u[1] = *(const uint4*)(p + 16);
      }
      #pragma unroll
      for (int mt = 0; mt < 2; ++mt)
        #pragma unroll
        for (int nt = 0; nt < 4; ++nt)
          sacc[mt][nt] = WMMA_BF16(aq[mt][kk].v, bk[nt].v, sacc[mt][nt]);
    }

    float mv[4];
    #pragma unroll
    for (int nt = 0; nt < 4; ++nt) mv[nt] = mp[j + nt * 16 + lrow] * -1e9f;

    // --- online softmax (row lives across the 16 lanes of one half-wave) ---
    #pragma unroll
    for (int mt = 0; mt < 2; ++mt)
      #pragma unroll
      for (int v = 0; v < 8; ++v) {
        float l[4];
        #pragma unroll
        for (int nt = 0; nt < 4; ++nt) l[nt] = sacc[mt][nt][v] * 0.125f + mv[nt];
        float bmax = fmaxf(fmaxf(l[0], l[1]), fmaxf(l[2], l[3]));
        #pragma unroll
        for (int off = 8; off >= 1; off >>= 1)
          bmax = fmaxf(bmax, __shfl_xor(bmax, off, 16));
        const float nm   = fmaxf(rmax[mt][v], bmax);
        const float corr = __expf(rmax[mt][v] - nm);
        rmax[mt][v] = nm;
        float ps = 0.0f;
        const int prow = mt * 16 + v + 8 * half;
        #pragma unroll
        for (int nt = 0; nt < 4; ++nt) {
          const float pe = __expf(l[nt] - nm);
          ps += pe;
          Pw[prow * LP + nt * 16 + lrow] = f32_to_bf16(pe);
        }
        #pragma unroll
        for (int off = 8; off >= 1; off >>= 1)
          ps += __shfl_xor(ps, off, 16);
        rsum[mt][v] = rsum[mt][v] * corr + ps;
        #pragma unroll
        for (int dn = 0; dn < 4; ++dn) acc_o[mt][dn][v] *= corr;
      }

    // --- O += P V (P re-read from per-wave LDS in A-fragment layout) ---
    #pragma unroll
    for (int kk = 0; kk < 2; ++kk) {
      FragU ap[2], bv4[4];
      #pragma unroll
      for (int mt = 0; mt < 2; ++mt) {
        const unsigned short* p = &Pw[(mt * 16 + lrow) * LP + kk * 32 + 8 * half];
        ap[mt].u[0] = *(const uint4*)p;
        ap[mt].u[1] = *(const uint4*)(p + 16);
      }
      #pragma unroll
      for (int dn = 0; dn < 4; ++dn) {
        const unsigned short* p = &Vs[(dn * 16 + lrow) * LV + kk * 32 + 8 * half];
        bv4[dn].u[0] = *(const uint4*)p;
        bv4[dn].u[1] = *(const uint4*)(p + 16);
      }
      #pragma unroll
      for (int mt = 0; mt < 2; ++mt)
        #pragma unroll
        for (int dn = 0; dn < 4; ++dn)
          acc_o[mt][dn] = WMMA_BF16(ap[mt].v, bv4[dn].v, acc_o[mt][dn]);
    }
    __syncthreads();
  }

  #pragma unroll
  for (int mt = 0; mt < 2; ++mt)
    #pragma unroll
    for (int dn = 0; dn < 4; ++dn)
      #pragma unroll
      for (int v = 0; v < 8; ++v) {
        const int row = q0 + mt * 16 + v + 8 * half;
        const int col = h * DH + dn * 16 + lrow;
        const float val = acc_o[mt][dn][v] / rsum[mt][v];
        Ob[(size_t)(b * S + row) * D + col] = f32_to_bf16(val);
      }
}

// ---------------------------------------------------------------------------
extern "C" void kernel_launch(void* const* d_in, const int* in_sizes, int n_in,
                              void* d_out, int out_size, void* d_ws, size_t ws_size,
                              hipStream_t stream) {
  (void)in_sizes; (void)n_in; (void)out_size; (void)ws_size;

  const float* query = (const float*)d_in[0];
  const float* key   = (const float*)d_in[1];
  const float* value = (const float*)d_in[2];
  const float* maskp = (const float*)d_in[3];
  const float* Wq = (const float*)d_in[4];  const float* bq = (const float*)d_in[5];
  const float* Wk = (const float*)d_in[6];  const float* bk = (const float*)d_in[7];
  const float* Wv = (const float*)d_in[8];  const float* bv = (const float*)d_in[9];
  const float* Wo = (const float*)d_in[10]; const float* bo = (const float*)d_in[11];

  const size_t NQ = (size_t)2 * 2048 * 1024;   // B*S*D
  const size_t NW = (size_t)1024 * 1024;       // D*D
  unsigned short* ws = (unsigned short*)d_ws;  // 64 MB bf16 scratch total
  unsigned short* xq  = ws;
  unsigned short* xk  = xq + NQ;
  unsigned short* xv  = xk + NQ;
  unsigned short* wtq = xv + NQ;
  unsigned short* wtk = wtq + NW;
  unsigned short* wtv = wtk + NW;
  unsigned short* wto = wtv + NW;
  unsigned short* q_ws = wto + NW;
  unsigned short* k_ws = q_ws + NQ;
  unsigned short* v_ws = k_ws + NQ;
  unsigned short* a_ws = v_ws + NQ;

  const int n4 = (int)(NQ / 4);
  cvt_bf16_kernel<<<n4 / 256, 256, 0, stream>>>(query, xq, n4);
  cvt_bf16_kernel<<<n4 / 256, 256, 0, stream>>>(key,   xk, n4);
  cvt_bf16_kernel<<<n4 / 256, 256, 0, stream>>>(value, xv, n4);

  dim3 wtg(32, 32), wtb(32, 8);
  wt_kernel<<<wtg, wtb, 0, stream>>>(Wq, wtq);
  wt_kernel<<<wtg, wtb, 0, stream>>>(Wk, wtk);
  wt_kernel<<<wtg, wtb, 0, stream>>>(Wv, wtv);
  wt_kernel<<<wtg, wtb, 0, stream>>>(Wo, wto);

  dim3 gg(1024 / 128, 4096 / 128);             // (N tiles, M tiles)
  gemm_bias_kernel<false><<<gg, 256, 0, stream>>>(xq, wtq, bq, q_ws);
  gemm_bias_kernel<false><<<gg, 256, 0, stream>>>(xk, wtk, bk, k_ws);
  gemm_bias_kernel<false><<<gg, 256, 0, stream>>>(xv, wtv, bv, v_ws);

  flash_attn_kernel<<<dim3(32, 8), 256, 0, stream>>>(q_ws, k_ws, v_ws, maskp, a_ws);

  gemm_bias_kernel<true><<<gg, 256, 0, stream>>>(a_ws, wto, bo, (float*)d_out);
}
